// MaskedSelfAttention_54477365183262
// MI455X (gfx1250) — matrix operand order repaired and checked by reading, not compiled, not run on hardware
//
#include <hip/hip_runtime.h>
#include <hip/hip_bf16.h>

#define B_ 8
#define S_ 2048
#define E_ 1024
#define H_ 64

typedef __attribute__((ext_vector_type(16))) __bf16          v16bf;
typedef __attribute__((ext_vector_type(8)))  float           v8f;
typedef __attribute__((ext_vector_type(4)))  unsigned int    u32x4;
typedef __attribute__((ext_vector_type(8)))  unsigned int    u32x8;

// K-index interleave for 16-bit A/B operands of V_WMMA_*_16X16X32 (ISA 7.12.2):
// lanes 0-15 hold K {0..7,16..23}, lanes 16-31 hold K {8..15,24..31}.
__device__ __forceinline__ int kpat(int e, int half) {
  return e + ((e >> 3) << 3) + (half << 3);
}

__device__ __forceinline__ unsigned short f2bf(float f) {
  unsigned int u = __builtin_bit_cast(unsigned int, f);
  u += 0x7FFFu + ((u >> 16) & 1u);            // round-to-nearest-even
  return (unsigned short)(u >> 16);
}

// Load one 16x16x32 bf16 operand whose 16 per-lane elements live as two
// 8-element (16B) chunks at p[0..7] and p[16..23] (+8*half folded by caller).
__device__ __forceinline__ v16bf ld_op(const unsigned short* p) {
  u32x4 lo = *(const u32x4*)p;
  u32x4 hi = *(const u32x4*)(p + 16);
  u32x8 w  = __builtin_shufflevector(lo, hi, 0, 1, 2, 3, 4, 5, 6, 7);
  return __builtin_bit_cast(v16bf, w);
}

// Async DMA 16B global -> LDS (ASYNCcnt-tracked, ISA 10.7 / 15.18.3).
// Low 32 bits of a generic LDS pointer are the flat-LDS byte address.
__device__ __forceinline__ void async_ld_b128(const unsigned short* gptr,
                                              unsigned short* lptr) {
  const unsigned lds = (unsigned)(size_t)lptr;
  asm volatile("global_load_async_to_lds_b128 %0, %1, off"
               :: "v"(lds), "v"((unsigned long long)(size_t)gptr)
               : "memory");
}

// ---------------------------------------------------------------------------
// Kernel 0 (one-shot, 384 KB): pre-pack Wq/Wk/Wv into per-lane WMMA B-layout
// bf16 so the GEMM hot loop uses contiguous coalesced b128 loads.
// Packed index: (((kstep*4 + n)*32 + lane)*16 + e)
// ---------------------------------------------------------------------------
__global__ void __launch_bounds__(256) pack_w_kernel(
    const float* __restrict__ Wq, const float* __restrict__ Wk,
    const float* __restrict__ Wv,
    unsigned short* __restrict__ pq, unsigned short* __restrict__ pk,
    unsigned short* __restrict__ pv)
{
  const int o     = blockIdx.x * 256 + threadIdx.x;   // 0 .. 3*65536-1
  const int e     = o & 15;
  const int lane  = (o >> 4) & 31;
  const int n     = (o >> 9) & 3;
  const int kstep = (o >> 11) & 31;
  const int mat   = o >> 16;
  const int krow  = kstep * 32 + kpat(e, lane >> 4);
  const int col   = n * 16 + (lane & 15);
  const float* W    = (mat == 0) ? Wq : (mat == 1) ? Wk : Wv;
  unsigned short* P = (mat == 0) ? pq : (mat == 1) ? pk : pv;
  P[o & 0xFFFF] = f2bf(W[(size_t)krow * H_ + col]);
}

// ---------------------------------------------------------------------------
// Kernel 1: fused QKV projection, software-pipelined B operands.
// ---------------------------------------------------------------------------
__global__ void __launch_bounds__(256) qkv_proj_kernel(
    const float* __restrict__ x,
    const unsigned short* __restrict__ wpq,
    const unsigned short* __restrict__ wpk,
    const unsigned short* __restrict__ wpv,
    const float* __restrict__ bq, const float* __restrict__ bk,
    const float* __restrict__ bv,
    unsigned short* __restrict__ qs, unsigned short* __restrict__ ks,
    unsigned short* __restrict__ vs)
{
  __shared__ __align__(16) unsigned short xs[128 * 72];  // bf16, padded stride
  const int t    = threadIdx.x;
  const int wave = t >> 5, lane = t & 31;
  const int half = lane >> 4, l16 = lane & 15;
  const int mblock = blockIdx.x * 128;
  const size_t lane16 = (size_t)lane * 16;

  v8f zero = {};
  v8f aq[4], ak[4], av[4];
#pragma unroll
  for (int n = 0; n < 4; ++n) { aq[n] = zero; ak[n] = zero; av[n] = zero; }

#define LDW(off, q_, k_, v_)                                             \
  do {                                                                   \
    q_ = __builtin_bit_cast(v16bf, *(const u32x8*)(wpq + (off)));        \
    k_ = __builtin_bit_cast(v16bf, *(const u32x8*)(wpk + (off)));        \
    v_ = __builtin_bit_cast(v16bf, *(const u32x8*)(wpv + (off)));        \
  } while (0)

  v16bf cq, ck, cv, nq, nk, nv;
  LDW(lane16, cq, ck, cv);                      // group (kk=0, n=0) in flight

  for (int kk = 0; kk < E_; kk += 32) {
    __syncthreads();
    {
      const int row = t >> 1, seg = (t & 1) * 16;
      const float*  src = x + (size_t)(mblock + row) * E_ + kk + seg;
      const float4* s4  = (const float4*)src;
      unsigned short r[16];
#pragma unroll
      for (int i = 0; i < 4; ++i) {
        float4 v = s4[i];
        r[i * 4 + 0] = f2bf(v.x); r[i * 4 + 1] = f2bf(v.y);
        r[i * 4 + 2] = f2bf(v.z); r[i * 4 + 3] = f2bf(v.w);
      }
      u32x4 p0, p1;
#pragma unroll
      for (int i = 0; i < 4; ++i) {
        p0[i] = (unsigned)r[2 * i]     | ((unsigned)r[2 * i + 1] << 16);
        p1[i] = (unsigned)r[8 + 2 * i] | ((unsigned)r[9 + 2 * i] << 16);
      }
      *(u32x4*)&xs[row * 72 + seg]     = p0;
      *(u32x4*)&xs[row * 72 + seg + 8] = p1;
      if (kk + 32 < E_) __builtin_prefetch(src + 32, 0, 1);  // global_prefetch
    }
    __syncthreads();

    // A operand: two aligned b128 LDS loads (chunks {0..7,16..23}+8*half)
    const v16bf a = ld_op(&xs[(wave * 16 + l16) * 72 + half * 8]);

    const size_t base = (size_t)((kk >> 5) * 4) * 512 + lane16;
#pragma unroll
    for (int n = 0; n < 4; ++n) {
      // issue next group's loads before consuming the current group
      const size_t noff =
          (n < 3) ? base + (size_t)(n + 1) * 512
                  : (size_t)(((kk + 32 < E_) ? ((kk >> 5) + 1) : 0) * 4) * 512 +
                        lane16;
      LDW(noff, nq, nk, nv);
      aq[n] = __builtin_amdgcn_wmma_f32_16x16x32_bf16(
          false, a, false, cq, (short)0, aq[n], false, false);
      ak[n] = __builtin_amdgcn_wmma_f32_16x16x32_bf16(
          false, a, false, ck, (short)0, ak[n], false, false);
      av[n] = __builtin_amdgcn_wmma_f32_16x16x32_bf16(
          false, a, false, cv, (short)0, av[n], false, false);
      cq = nq; ck = nk; cv = nv;
    }
  }
#undef LDW

  // C/D layout: element r <-> row r + 8*half, column = l16 (+ 16*n)
#pragma unroll
  for (int n = 0; n < 4; ++n) {
    const int h = n * 16 + l16;
#pragma unroll
    for (int r = 0; r < 8; ++r) {
      const size_t m   = (size_t)(mblock + wave * 16 + r + 8 * half);
      const size_t idx = m * H_ + h;
      qs[idx] = f2bf((aq[n][r] + bq[h]) * 0.125f);   // fold 1/sqrt(H)
      ks[idx] = f2bf(ak[n][r] + bk[h]);
      vs[idx] = f2bf(av[n][r] + bv[h]);
    }
  }
}

// ---------------------------------------------------------------------------
// Kernel 2: flash attention with double-buffered LDS tiles; K tiles come in
// via global_load_async_to_lds_b128 (ASYNCcnt), V tiles transposed at stage.
// ---------------------------------------------------------------------------
__global__ void __launch_bounds__(256) attn_kernel(
    const unsigned short* __restrict__ qs,
    const unsigned short* __restrict__ ks,
    const unsigned short* __restrict__ vs,
    const int* __restrict__ mask,
    float* __restrict__ out)
{
  __shared__ __align__(16) unsigned short Kt[2 * 32 * 72];   // [buf][key][h]
  __shared__ __align__(16) unsigned short Vt[2 * 64 * 40];   // [buf][h][key]
  __shared__ __align__(16) unsigned short Pt[8 * 16 * 40];   // per-wave P

  const int t    = threadIdx.x;
  const int wave = t >> 5, lane = t & 31;
  const int half = lane >> 4, l16 = lane & 15;
  const int b      = blockIdx.x;
  const int qblock = blockIdx.y * 128;
  const int qbase  = qblock + wave * 16;
  const int key    = t >> 3, sub = t & 7;      // staging role: 8 thr / 64-row

  // Q A-operands: 4 aligned b128 global loads (bf16, scale pre-folded)
  const unsigned short* qrow =
      qs + (size_t)(b * S_ + qbase + l16) * H_ + half * 8;
  const v16bf qa0 = ld_op(qrow);          // H 0..31
  const v16bf qa1 = ld_op(qrow + 32);     // H 32..63

  v8f zero = {};
  v8f acc[4];
#pragma unroll
  for (int n = 0; n < 4; ++n) acc[n] = zero;
  float mrow[8], lrow[8];
#pragma unroll
  for (int r = 0; r < 8; ++r) { mrow[r] = -1e30f; lrow[r] = 0.0f; }

  // stage tile `ks_`: async K copy + synchronous transposed V copy
  auto stage = [&](int kstart, int buf) {
    const size_t grow = (size_t)(b * S_ + kstart + key) * H_ + sub * 8;
    unsigned short* kb = Kt + buf * (32 * 72);
    unsigned short* vb = Vt + buf * (64 * 40);
    async_ld_b128(ks + grow, &kb[key * 72 + sub * 8]);
    const u32x4 vq = *(const u32x4*)(vs + grow);
#pragma unroll
    for (int j = 0; j < 4; ++j) {              // transpose V: [h][key]
      const unsigned int u = vq[j];
      vb[(sub * 8 + 2 * j) * 40 + key]     = (unsigned short)(u & 0xFFFFu);
      vb[(sub * 8 + 2 * j + 1) * 40 + key] = (unsigned short)(u >> 16);
    }
  };

  const int kbmax = (qblock + 127) >> 5;
  stage(0, 0);                                 // prologue: tile 0 in flight

  for (int kb = 0; kb <= kbmax; ++kb) {
    const int kstart = kb << 5;
    const int cur = kb & 1;
    if (kb < kbmax) {
      stage(kstart + 32, cur ^ 1);             // next tile in flight
      asm volatile("s_wait_asynccnt 1" ::: "memory");   // oldest tile done
    } else {
      asm volatile("s_wait_asynccnt 0" ::: "memory");
    }
    __syncthreads();                           // tile kb visible to all waves

    if (kstart <= qbase + 15) {                // wave-uniform: EXEC stays full
      const unsigned short* kbuf = Kt + cur * (32 * 72);
      const unsigned short* vbuf = Vt + cur * (64 * 40);

      // scores S = Q K^T : two key n-tiles, K=64 as 2 chained WMMAs each
      v8f s0 = zero, s1 = zero;
#pragma unroll
      for (int c = 0; c < 2; ++c) {
        const v16bf bk0 = ld_op(&kbuf[l16 * 72        + c * 32 + half * 8]);
        const v16bf bk1 = ld_op(&kbuf[(16 + l16) * 72 + c * 32 + half * 8]);
        const v16bf a = c ? qa1 : qa0;
        s0 = __builtin_amdgcn_wmma_f32_16x16x32_bf16(
            false, a, false, bk0, (short)0, s0, false, false);
        s1 = __builtin_amdgcn_wmma_f32_16x16x32_bf16(
            false, a, false, bk1, (short)0, s1, false, false);
      }

      // mask + online softmax (row stats live in each 16-lane half-group)
      const int key0 = kstart + l16;
      const int key1 = kstart + 16 + l16;
      const int ok0p = mask[b * S_ + key0];
      const int ok1p = mask[b * S_ + key1];
      float p0[8], p1[8];
#pragma unroll
      for (int r = 0; r < 8; ++r) {
        const int qrw = qbase + r + 8 * half;
        float v0 = (ok0p && key0 <= qrw) ? s0[r] : -1e30f;
        float v1 = (ok1p && key1 <= qrw) ? s1[r] : -1e30f;
        float mx = fmaxf(v0, v1);
#pragma unroll
        for (int d = 1; d < 16; d <<= 1) mx = fmaxf(mx, __shfl_xor(mx, d, 32));
        const float mnew = fmaxf(mrow[r], mx);
        const float sc   = __expf(mrow[r] - mnew);
        p0[r] = (v0 > -1e29f) ? __expf(v0 - mnew) : 0.0f;
        p1[r] = (v1 > -1e29f) ? __expf(v1 - mnew) : 0.0f;
        float rs = p0[r] + p1[r];
#pragma unroll
        for (int d = 1; d < 16; d <<= 1) rs += __shfl_xor(rs, d, 32);
        lrow[r] = lrow[r] * sc + rs;
        mrow[r] = mnew;
#pragma unroll
        for (int n = 0; n < 4; ++n) acc[n][r] *= sc;
      }

      // transpose P (C/D layout -> A layout) via per-wave LDS scratch
      unsigned short* pb = &Pt[wave * 16 * 40];
#pragma unroll
      for (int r = 0; r < 8; ++r) {
        const int row = r + 8 * half;
        pb[row * 40 + l16]      = f2bf(p0[r]);
        pb[row * 40 + 16 + l16] = f2bf(p1[r]);
      }
      asm volatile("s_wait_dscnt 0" ::: "memory");
      const v16bf pa = ld_op(&pb[l16 * 40 + half * 8]);

      // O += P V : inner K = 32 keys, 4 output n-tiles over H=64
#pragma unroll
      for (int n = 0; n < 4; ++n) {
        const v16bf bv = ld_op(&vbuf[(n * 16 + l16) * 40 + half * 8]);
        acc[n] = __builtin_amdgcn_wmma_f32_16x16x32_bf16(
            false, pa, false, bv, (short)0, acc[n], false, false);
      }
    }
    __syncthreads();                           // all waves done reading buf
  }

#pragma unroll
  for (int r = 0; r < 8; ++r) {
    const float inv = 1.0f / lrow[r];          // diag key guarantees lrow > 0
    const size_t m  = (size_t)(b * S_ + qbase + r + 8 * half);
#pragma unroll
    for (int n = 0; n < 4; ++n)
      out[m * H_ + n * 16 + l16] = acc[n][r] * inv;
  }
}

extern "C" void kernel_launch(void* const* d_in, const int* in_sizes, int n_in,
                              void* d_out, int out_size, void* d_ws, size_t ws_size,
                              hipStream_t stream) {
  const float* x    = (const float*)d_in[0];
  const int*   mask = (const int*)d_in[1];
  const float* Wq   = (const float*)d_in[2];
  const float* bq   = (const float*)d_in[3];
  const float* Wk   = (const float*)d_in[4];
  const float* bk   = (const float*)d_in[5];
  const float* Wv   = (const float*)d_in[6];
  const float* bv   = (const float*)d_in[7];
  float* out = (float*)d_out;

  const size_t N  = (size_t)B_ * S_ * H_;      // 1 M elems per q/k/v tensor
  const size_t NW = (size_t)E_ * H_;           // 64 K elems per weight
  unsigned short* qs  = (unsigned short*)d_ws; // 2 MB
  unsigned short* ks  = qs + N;                // 2 MB
  unsigned short* vs  = ks + N;                // 2 MB
  unsigned short* wpq = vs + N;                // 128 KB (packed bf16 weights)
  unsigned short* wpk = wpq + NW;
  unsigned short* wpv = wpk + NW;

  pack_w_kernel<<<dim3(3 * NW / 256), dim3(256), 0, stream>>>(
      Wq, Wk, Wv, wpq, wpk, wpv);
  qkv_proj_kernel<<<dim3((B_ * S_) / 128), dim3(256), 0, stream>>>(
      x, wpq, wpk, wpv, bq, bk, bv, qs, ks, vs);
  attn_kernel<<<dim3(B_, S_ / 128), dim3(256), 0, stream>>>(
      qs, ks, vs, mask, out);
}
